// SwinTransformer_76647986364433
// MI455X (gfx1250) — compile-verified, matrix operand order checked by reading, not run on hardware
//
#include <hip/hip_runtime.h>
#include <hip/hip_bf16.h>

// ---------------------------------------------------------------------------
// Swin window attention for MI455X (gfx1250, wave32, WMMA bf16 w/ f32 accum,
// Tensor Data Mover for bf16 tile staging).
//   x:[32,56,56,256] f32 ; Wq/Wk/Wv:[256,256] ; bq/bk/bv:[256]
//   out:[32,64,49,256] f32
// ---------------------------------------------------------------------------

#define C_DIM    256
#define TOKENS   100352      // 32 * 64 * 49
#define NWIN     2048        // 32 * 64
#define HEADS    8
#define HDIM     32

typedef __attribute__((ext_vector_type(16))) __bf16 v16bf;
typedef __attribute__((ext_vector_type(8)))  __bf16 v8bf;
typedef __attribute__((ext_vector_type(8)))  float  v8f;
typedef __attribute__((ext_vector_type(4)))  unsigned int u32x4;
typedef __attribute__((ext_vector_type(8)))  int    i32x8;
typedef __attribute__((ext_vector_type(4)))  int    i32x4;

__device__ __forceinline__ v8f wmma_bf16(v16bf a, v16bf b, v8f c) {
    return __builtin_amdgcn_wmma_f32_16x16x32_bf16(
        /*neg_a=*/false, a, /*neg_b=*/false, b,
        /*c_mod=*/(short)0, c, /*reuse_a=*/false, /*reuse_b=*/false);
}

// Load one 16x32 bf16 fragment (A-layout; B uses the same pattern on B^T
// stored row-major) from LDS.  Per CDNA5 ISA 7.12.2:
//   lanes 0-15 : row M=lane,    K = {0..7} in v0-3,  {16..23} in v4-7
//   lanes 16-31: row M=lane-16, K = {8..15} in v0-3, {24..31} in v4-7
__device__ __forceinline__ v16bf load_frag(const __bf16* base, int stride, int lane) {
    const int m  = lane & 15;
    const int kb = (lane & 16) ? 8 : 0;
    const __bf16* p = base + m * stride + kb;
    v8bf lo = *(const v8bf*)(p);        // K = kb .. kb+7      (16B aligned)
    v8bf hi = *(const v8bf*)(p + 16);   // K = kb+16 .. kb+23
    return __builtin_shufflevector(lo, hi, 0,1,2,3,4,5,6,7,8,9,10,11,12,13,14,15);
}

// Generic (flat) LDS pointer -> byte offset within workgroup LDS allocation.
// On amdgcn the flat form of an LDS address carries the offset in addr[31:0].
__device__ __forceinline__ unsigned lds_offset(const void* p) {
    return (unsigned)(unsigned long long)(uintptr_t)p;
}

// Issue a 2D TENSOR_LOAD_TO_LDS of bf16 data (D# per CDNA5 ISA ch.8):
//   tile_w elements per row (contiguous in memory and LDS),
//   tile_h rows, row stride `stride` elements in global memory.
__device__ __forceinline__ void tdm_load_2d_bf16(unsigned lds_off,
                                                 unsigned long long gaddr,
                                                 unsigned tile_w, unsigned tile_h,
                                                 unsigned long long stride) {
    union { unsigned long long q[2]; u32x4 v; } g0;
    // group0: count=1 | lds_addr[63:32] ; global_addr[120:64] ; type[127:126]=2
    g0.q[0] = 1ull | ((unsigned long long)lds_off << 32);
    g0.q[1] = (gaddr & 0x01FFFFFFFFFFFFFFull) | (2ull << 62);

    union { unsigned long long q[4]; i32x8 v; } g1;
    // group1: data_size[17:16]=1 (2B) ; tensor_dim0[79:48] ; tensor_dim1[111:80]
    //         tile_dim0[127:112] ; tile_dim1[143:128] ; dim0_stride[207:160]
    g1.q[0] = (1ull << 16) | (((unsigned long long)(tile_w & 0xFFFF)) << 48);
    g1.q[1] = ((unsigned long long)(tile_w >> 16) & 0xFFFF)
            | ((unsigned long long)tile_h << 16)
            | (((unsigned long long)(tile_w & 0xFFFF)) << 48);
    g1.q[2] = (unsigned long long)(tile_h & 0xFFFF)
            | ((stride & 0xFFFFFFFFull) << 32);
    g1.q[3] = (stride >> 32) & 0xFFFFull;

    i32x4 z4 = {0, 0, 0, 0};
#if defined(__clang_major__) && (__clang_major__ >= 23)
    i32x8 z8 = {0, 0, 0, 0, 0, 0, 0, 0};
    __builtin_amdgcn_tensor_load_to_lds(g0.v, g1.v, z4, z4, z8, 0);
#else
    __builtin_amdgcn_tensor_load_to_lds(g0.v, g1.v, z4, z4, 0);
#endif
}

// ---------------------------------------------------------------------------
// Kernel 1: fp32 -> bf16 weight conversion.  Wb layout: [3][256][256], same
// [d][c] order as the inputs (this IS B^T row-major for y = x @ W^T).
// ---------------------------------------------------------------------------
__global__ __launch_bounds__(256) void cvt_w(const float* __restrict__ Wq,
                                             const float* __restrict__ Wk,
                                             const float* __restrict__ Wv,
                                             __bf16* __restrict__ Wb) {
    int idx = blockIdx.x * 256 + threadIdx.x;         // 0 .. 3*65536-1
    if (idx >= 3 * 65536) return;
    int m = idx >> 16;
    int o = idx & 65535;
    const float* src = (m == 0) ? Wq : (m == 1) ? Wk : Wv;
    Wb[idx] = (__bf16)src[o];
}

// ---------------------------------------------------------------------------
// Kernel 2: QKV projection GEMM, q/k/v fused per workgroup.
//   grid = (1568, 4) -> (M-tile of 64 rows, N-tile of 64 cols); z folded in.
//   block = 128 threads (4 waves).  B panels ([3][64][256] bf16, 96KB LDS)
//   are loaded once per block by the TDM; the A tile (f32 gather + cvt) is
//   staged per 32-wide K slice and feeds 12 WMMAs per iteration.
// ---------------------------------------------------------------------------
__global__ __launch_bounds__(128) void qkv_gemm(const float* __restrict__ x,
                                                const __bf16* __restrict__ Wb,
                                                const float* __restrict__ bq,
                                                const float* __restrict__ bk,
                                                const float* __restrict__ bv,
                                                __bf16* __restrict__ qkv) {
    __shared__ __bf16 Bp[3 * 64 * 256];   // 96 KB: B panels for q,k,v
    __shared__ __bf16 As[64 * 32];        //  4 KB: A K-slice

    const int tid  = threadIdx.x;
    const int wave = tid >> 5;
    const int lane = tid & 31;
    const int m0   = blockIdx.x * 64;
    const int n0   = blockIdx.y * 64;

    // --- TDM: stage the three 64x256 bf16 weight panels into LDS -----------
    if (tid == 0) {
#pragma unroll
        for (int z = 0; z < 3; ++z) {
            tdm_load_2d_bf16(lds_offset(&Bp[z << 14]),
                             (unsigned long long)(uintptr_t)(Wb + ((long)z << 16) +
                                                             (long)n0 * C_DIM),
                             /*tile_w=*/C_DIM, /*tile_h=*/64, /*stride=*/C_DIM);
        }
    }

    // staging assignment: each thread owns (row lr, 16-wide K half lk)
    const int lr = tid >> 1;            // 0..63
    const int lk = (tid & 1) << 4;      // 0 or 16

    // window-partition gather address for this A row
    const int row = m0 + lr;
    const int t   = row % 49;
    const int w   = (row / 49) & 63;
    const int b   = row / (49 * 64);
    const long xoff = ((long)(b * 56 + (w >> 3) * 7 + t / 7) * 56 +
                       ((w & 7) * 7 + t % 7)) * C_DIM;
    const float* arow = x + xoff + lk;

    v8f acc[3][4];
#pragma unroll
    for (int z = 0; z < 3; ++z)
#pragma unroll
        for (int j = 0; j < 4; ++j)
#pragma unroll
            for (int g = 0; g < 8; ++g) acc[z][j][g] = 0.0f;

    __builtin_amdgcn_s_wait_tensorcnt(0);
    __syncthreads();

    for (int kk = 0; kk < C_DIM; kk += 32) {
        // stage A: 16 f32 -> 16 bf16 into LDS
        const float4* ap = (const float4*)(arow + kk);
        float4 f0 = ap[0], f1 = ap[1], f2 = ap[2], f3 = ap[3];
        __bf16* ad = &As[(lr << 5) + lk];
        ad[0]  = (__bf16)f0.x; ad[1]  = (__bf16)f0.y; ad[2]  = (__bf16)f0.z; ad[3]  = (__bf16)f0.w;
        ad[4]  = (__bf16)f1.x; ad[5]  = (__bf16)f1.y; ad[6]  = (__bf16)f1.z; ad[7]  = (__bf16)f1.w;
        ad[8]  = (__bf16)f2.x; ad[9]  = (__bf16)f2.y; ad[10] = (__bf16)f2.z; ad[11] = (__bf16)f2.w;
        ad[12] = (__bf16)f3.x; ad[13] = (__bf16)f3.y; ad[14] = (__bf16)f3.z; ad[15] = (__bf16)f3.w;

        if (kk + 32 < C_DIM)                    // gfx1250 global_prefetch_b8
            __builtin_prefetch(arow + kk + 32, 0, 0);
        __syncthreads();

        v16bf a = load_frag(&As[wave << 9], 32, lane);
#pragma unroll
        for (int z = 0; z < 3; ++z) {
#pragma unroll
            for (int j = 0; j < 4; ++j) {
                v16bf bf = load_frag(&Bp[(z << 14) + (j << 4) * C_DIM + kk],
                                     C_DIM, lane);
                acc[z][j] = wmma_bf16(a, bf, acc[z][j]);
            }
        }
        __syncthreads();
    }

    // epilogue: bias add, store bf16.  C/D layout: vgpr g, lanes 0-15 row g,
    // lanes 16-31 row g+8; col = lane&15 within 16-wide N tile.
    const int nlo  = lane & 15;
    const int rsel = (lane & 16) ? 8 : 0;
#pragma unroll
    for (int z = 0; z < 3; ++z) {
        const float* bias = (z == 0) ? bq : (z == 1) ? bk : bv;
#pragma unroll
        for (int j = 0; j < 4; ++j) {
            const int n = n0 + (j << 4) + nlo;
            const float bb = bias[n];
#pragma unroll
            for (int g = 0; g < 8; ++g) {
                const int r = m0 + (wave << 4) + rsel + g;
                qkv[((long)z * TOKENS + r) * C_DIM + n] = (__bf16)(acc[z][j][g] + bb);
            }
        }
    }
}

// ---------------------------------------------------------------------------
// Kernel 3: per-(window, head) attention.  grid = (2048, 8), block = 128.
//   S = Q K^T * (1/16), masked softmax over 49 valid cols, O = P V.
//   T padded 49 -> 64.  Q/K head slices staged by TDM (2D tile 32x49,
//   stride 256); V transposed manually for the B-fragment layout.
// ---------------------------------------------------------------------------
__global__ __launch_bounds__(128) void win_attn(const __bf16* __restrict__ qkv,
                                                float* __restrict__ out) {
    __shared__ __bf16 qs[64 * 32];   // [token][c]
    __shared__ __bf16 ks[64 * 32];   // [token][c]
    __shared__ __bf16 vt[32 * 64];   // [c][token]  (V^T for B fragments)
    __shared__ __bf16 ps[64 * 64];   // [row][col]  softmax(P) as bf16

    const int tid  = threadIdx.x;
    const int wave = tid >> 5;
    const int lane = tid & 31;
    const int win  = blockIdx.x;          // 0..2047  (= b*64 + w)
    const int head = blockIdx.y;          // 0..7
    const long r0  = (long)win * 49;
    const int  cb  = head * HDIM;

    const __bf16* qg = qkv;
    const __bf16* kg = qkv + (long)TOKENS * C_DIM;
    const __bf16* vg = qkv + (long)2 * TOKENS * C_DIM;

    // --- TDM: stage q/k head slices (49 rows x 32 bf16, stride 256) --------
    if (tid == 0) {
        tdm_load_2d_bf16(lds_offset(qs),
                         (unsigned long long)(uintptr_t)(qg + r0 * C_DIM + cb),
                         /*tile_w=*/HDIM, /*tile_h=*/49, /*stride=*/C_DIM);
        tdm_load_2d_bf16(lds_offset(ks),
                         (unsigned long long)(uintptr_t)(kg + r0 * C_DIM + cb),
                         /*tile_w=*/HDIM, /*tile_h=*/49, /*stride=*/C_DIM);
    }

    // zero the padded rows/cols (disjoint from the TDM-written region)
    for (int idx = tid; idx < 64 * 32; idx += 128) {
        if ((idx >> 5) >= 49) { qs[idx] = (__bf16)0.0f; ks[idx] = (__bf16)0.0f; }
    }
    for (int idx = tid; idx < 32 * 64; idx += 128) {
        if ((idx & 63) >= 49) vt[idx] = (__bf16)0.0f;
    }
    // manual transposed load of V head slice
    for (int idx = tid; idx < 49 * 32; idx += 128) {
        const int tt = idx >> 5, c = idx & 31;
        vt[(c << 6) + tt] = vg[(r0 + tt) * C_DIM + cb + c];
    }
    __builtin_amdgcn_s_wait_tensorcnt(0);
    __syncthreads();

    // ---- S = Q K^T : each wave does 16 rows x 64 cols, K = 32 ----
    v8f s[4];
#pragma unroll
    for (int j = 0; j < 4; ++j)
#pragma unroll
        for (int g = 0; g < 8; ++g) s[j][g] = 0.0f;

    {
        v16bf a = load_frag(&qs[wave << 9], 32, lane);
#pragma unroll
        for (int j = 0; j < 4; ++j) {
            v16bf bf = load_frag(&ks[j << 9], 32, lane);
            s[j] = wmma_bf16(a, bf, s[j]);
        }
    }

    // ---- masked softmax over each row (cols 0..48 valid) ----
    const float scale = 0.0625f;          // 1/sqrt(256)
    const int   nlo   = lane & 15;
    const int   rsel  = (lane & 16) ? 8 : 0;
#pragma unroll
    for (int g = 0; g < 8; ++g) {
        float vals[4];
        float m = -3.0e38f;
#pragma unroll
        for (int j = 0; j < 4; ++j) {
            const int col = (j << 4) + nlo;
            float v = s[j][g] * scale;
            if (col >= 49) v = -3.0e38f;
            vals[j] = v;
            m = fmaxf(m, v);
        }
        // reduce across the 16 lanes holding this row's columns
#pragma unroll
        for (int off = 1; off < 16; off <<= 1)
            m = fmaxf(m, __shfl_xor(m, off, 32));
        float sum = 0.0f;
#pragma unroll
        for (int j = 0; j < 4; ++j) {
            float e = __expf(vals[j] - m);
            vals[j] = e;
            sum += e;
        }
#pragma unroll
        for (int off = 1; off < 16; off <<= 1)
            sum += __shfl_xor(sum, off, 32);
        const float inv = 1.0f / sum;
        const int prow = (wave << 4) + rsel + g;
#pragma unroll
        for (int j = 0; j < 4; ++j)
            ps[(prow << 6) + (j << 4) + nlo] = (__bf16)(vals[j] * inv);
    }
    __syncthreads();

    // ---- O = P V : 16 rows x 32 cols per wave, K = 64 (two 32-steps) ----
    v8f o[2];
#pragma unroll
    for (int j = 0; j < 2; ++j)
#pragma unroll
        for (int g = 0; g < 8; ++g) o[j][g] = 0.0f;

#pragma unroll
    for (int kt = 0; kt < 2; ++kt) {
        v16bf a = load_frag(&ps[(wave << 10) + (kt << 5)], 64, lane);
#pragma unroll
        for (int j = 0; j < 2; ++j) {
            v16bf bf = load_frag(&vt[(j << 10) + (kt << 5)], 64, lane);
            o[j] = wmma_bf16(a, bf, o[j]);
        }
    }

    // ---- store fp32 output rows t < 49 ----
#pragma unroll
    for (int j = 0; j < 2; ++j) {
        const int c = cb + (j << 4) + nlo;
#pragma unroll
        for (int g = 0; g < 8; ++g) {
            const int tt = (wave << 4) + rsel + g;
            if (tt < 49) out[(r0 + tt) * C_DIM + c] = o[j][g];
        }
    }
}

// ---------------------------------------------------------------------------
extern "C" void kernel_launch(void* const* d_in, const int* in_sizes, int n_in,
                              void* d_out, int out_size, void* d_ws, size_t ws_size,
                              hipStream_t stream) {
    const float* x  = (const float*)d_in[0];
    const float* Wq = (const float*)d_in[1];
    const float* bq = (const float*)d_in[2];
    const float* Wk = (const float*)d_in[3];
    const float* bk = (const float*)d_in[4];
    const float* Wv = (const float*)d_in[5];
    const float* bv = (const float*)d_in[6];
    float* out = (float*)d_out;

    // workspace: [3][256][256] bf16 weights, then [3][100352][256] bf16 qkv
    __bf16* Wb  = (__bf16*)d_ws;
    __bf16* qkv = Wb + 3 * 65536;

    cvt_w<<<(3 * 65536 + 255) / 256, 256, 0, stream>>>(Wq, Wk, Wv, Wb);

    dim3 gGemm(TOKENS / 64, C_DIM / 64);      // (1568, 4), q/k/v fused
    qkv_gemm<<<gGemm, 128, 0, stream>>>(x, Wb, bq, bk, bv, qkv);

    dim3 gAttn(NWIN, HEADS);                  // (2048, 8)
    win_attn<<<gAttn, 128, 0, stream>>>(qkv, out);
}
